// RelationExtractionModule_87471303950886
// MI455X (gfx1250) — matrix operand order repaired and checked by reading, not compiled, not run on hardware
//
#include <hip/hip_runtime.h>

#define B_ 16
#define E_ 512
#define T_ 512
#define H_ 1024
#define R_ 32
#define P_ 8192
#define KW 3

typedef __attribute__((ext_vector_type(16))) _Float16 v16h;
typedef __attribute__((ext_vector_type(8)))  float    v8f;

__device__ __forceinline__ v8f wmma_f16(v16h a, v16h b, v8f c) {
  // D = A(16x32 f16) * B(32x16 f16) + C(16x16 f32)
  return __builtin_amdgcn_wmma_f32_16x16x32_f16(
      /*neg_a=*/false, a, /*neg_b=*/false, b,
      /*c_mod=*/(short)0, c, /*reuse_a=*/false, /*reuse_b=*/false);
}

// ---------------------------------------------------------------------------
// Residual conv1d: out = relu(conv(xin) + bias) + xin
// xin layout (B,E,T) f32; weights pre-converted to f16 layout [tap][m][c].
// Per batch: GEMM M=E_out, N=T(16 cols/block), K=E_in per tap (3 taps).
// Block: 8 waves -> 128(M) x 16(N). The B tile (32 channels x 18-col halo,
// padded to 24 cols so staging is branch-free) is shared by all waves and
// staged once per K-chunk in LDS as f16.
// ---------------------------------------------------------------------------
__global__ __launch_bounds__(256) void conv_res_kernel(
    const float* __restrict__ xin, const _Float16* __restrict__ wch,
    const float* __restrict__ bias, float* __restrict__ out)
{
  __shared__ _Float16 Xs[24][32];   // cols t0-1..t0+16 used; 18..23 = pad

  const int tid  = threadIdx.x;
  const int lane = tid & 31;
  const int wave = tid >> 5;
  const int b    = blockIdx.z;
  const int t0   = blockIdx.x * 16;
  const int m0   = blockIdx.y * 128 + wave * 16;
  const int ln   = lane & 15;   // A row / B col within tile
  const int hi   = lane >> 4;   // lane half -> K sub-range

  v8f acc = {};
  const float* xb = xin + (size_t)b * E_ * T_;

  // Branch-free staging map: each thread owns cols j = sj, sj+8, sj+16 (<24)
  // for channel sc. Clamp/validity hoisted out of the K loop.
  const int sj = tid >> 5;
  const int sc = tid & 31;
  int  tcs[3];
  bool vld[3];
#pragma unroll
  for (int u = 0; u < 3; ++u) {
    const int t = t0 + sj + u * 8 - 1;
    tcs[u] = t < 0 ? 0 : (t >= T_ ? T_ - 1 : t);
    vld[u] = (t >= 0) && (t < T_);
  }

  for (int c0 = 0; c0 < E_; c0 += 32) {
    const float* src = xb + (size_t)(c0 + sc) * T_;
#pragma unroll
    for (int u = 0; u < 3; ++u) {
      const float v = src[tcs[u]];
      Xs[sj + u * 8][sc] = vld[u] ? (_Float16)v : (_Float16)0.f;
    }
    __syncthreads();

#pragma unroll
    for (int tap = 0; tap < KW; ++tap) {
      // A fragment: wch[tap][m0+ln][c0 + k], contiguous in c -> 2x b128
      v16h a;
      {
        uint4* ap = reinterpret_cast<uint4*>(&a);
        const _Float16* pw =
            wch + ((size_t)tap * E_ + m0 + ln) * E_ + c0 + hi * 8;
        ap[0] = *reinterpret_cast<const uint4*>(pw);
        ap[1] = *reinterpret_cast<const uint4*>(pw + 16);
      }
      // B fragment from LDS: Xs[ln+tap][hi*8 .. +7] and [16+hi*8 .. +7] -> 2x ds_load_b128
      v16h bf;
      {
        uint4* bp = reinterpret_cast<uint4*>(&bf);
        const _Float16* ps = &Xs[ln + tap][hi * 8];
        bp[0] = *reinterpret_cast<const uint4*>(ps);
        bp[1] = *reinterpret_cast<const uint4*>(ps + 16);
      }
      acc = wmma_f16(a, bf, acc);
    }
    __syncthreads();
  }

  // Store: relu(acc + bias[row]) + residual
  const int col = t0 + ln;
#pragma unroll
  for (int r = 0; r < 8; ++r) {
    const int row = m0 + hi * 8 + r;
    float v = acc[r] + bias[row];
    v = v > 0.f ? v : 0.f;
    const size_t idx = ((size_t)b * E_ + row) * T_ + col;
    out[idx] = v + xin[idx];
  }
}

// ---------------------------------------------------------------------------
// Span-max gather: maxed[p, e] = max_span(head) + max_span(tail), f16 output
// ---------------------------------------------------------------------------
__global__ __launch_bounds__(256) void spanmax_kernel(
    const float* __restrict__ x, const int* __restrict__ bidx,
    const int* __restrict__ hidx, const int* __restrict__ hspan,
    const int* __restrict__ tidx, const int* __restrict__ tspan,
    _Float16* __restrict__ maxed)
{
  const int p  = blockIdx.x;
  const int b  = bidx[p];
  const int h0 = hidx[p], hs = hspan[p];
  const int u0 = tidx[p], us = tspan[p];
  for (int e = threadIdx.x; e < E_; e += blockDim.x) {
    const float* row = x + ((size_t)b * E_ + e) * T_;
    float hm = -3.4e38f, tm = -3.4e38f;
    for (int s = 0; s < hs; ++s) hm = fmaxf(hm, row[h0 + s]);
    for (int s = 0; s < us; ++s) tm = fmaxf(tm, row[u0 + s]);
    maxed[(size_t)p * E_ + e] = (_Float16)(hm + tm);
  }
}

// ---------------------------------------------------------------------------
// MLP GEMM: out[p, n] = act( sum_k A[p,k] * Wt[n,k] + bias[n] )
// A (PxK) f16 row-major, Wt (NxK) f16 row-major (W.T applied).
// Block: 16 P-rows x (nwaves*16) cols. A tile double-buffered in LDS via the
// gfx1250 async global->LDS path (ASYNCcnt): DMA of chunk k+1 overlaps the
// WMMA of chunk k.
// ---------------------------------------------------------------------------
__global__ __launch_bounds__(256) void mlp_gemm_kernel(
    const _Float16* __restrict__ A, const _Float16* __restrict__ Wt,
    const float* __restrict__ bias, _Float16* __restrict__ out_h,
    float* __restrict__ out_f, int K, int N, int do_relu)
{
  __shared__ _Float16 As[2][16][32];

  const int tid    = threadIdx.x;
  const int lane   = tid & 31;
  const int wave   = tid >> 5;
  const int nwaves = blockDim.x >> 5;
  const int p0     = blockIdx.x * 16;
  const int n0     = (blockIdx.y * nwaves + wave) * 16;
  const int ln     = lane & 15;
  const int hi     = lane >> 4;

  const float bb = bias[n0 + ln];
  v8f acc = {};

  // Issue async global->LDS copies of a 16x32 f16 tile (256 dwords)
  auto stage = [&](int k0, int buf) {
    for (int i = tid; i < 256; i += blockDim.x) {
      const int r = i >> 4, kp = i & 15;
      const unsigned lds_b = (unsigned)(size_t)&As[buf][r][kp * 2];
      const unsigned* gsrc =
          reinterpret_cast<const unsigned*>(A + (size_t)(p0 + r) * K + k0) + kp;
      asm volatile("global_load_async_to_lds_b32 %0, %1, off"
                   :: "v"(lds_b), "v"(gsrc) : "memory");
    }
  };

  // Prologue: fill buffer 0
  stage(0, 0);
  asm volatile("s_wait_asynccnt 0x0" ::: "memory");
  __syncthreads();

#pragma unroll 2
  for (int k0 = 0; k0 < K; k0 += 32) {
    const int buf = (k0 >> 5) & 1;

    // Kick off DMA of the next chunk into the other buffer (overlaps WMMA)
    if (k0 + 32 < K) stage(k0 + 32, buf ^ 1);

    // A fragment from LDS (2x ds_load_b128)
    v16h a;
    {
      uint4* ap = reinterpret_cast<uint4*>(&a);
      const int kb = hi * 8;
      ap[0] = *reinterpret_cast<const uint4*>(&As[buf][ln][kb]);
      ap[1] = *reinterpret_cast<const uint4*>(&As[buf][ln][kb + 16]);
    }
    // B fragment: Wt[n0+ln, k0+hi*8 .. +7] and [k0+16+hi*8 .. +7] (2x global_load_b128)
    v16h bf;
    {
      uint4* bp = reinterpret_cast<uint4*>(&bf);
      const _Float16* pw = Wt + (size_t)(n0 + ln) * K + k0 + hi * 8;
      bp[0] = *reinterpret_cast<const uint4*>(pw);
      bp[1] = *reinterpret_cast<const uint4*>(pw + 16);
    }

    acc = wmma_f16(a, bf, acc);

    // Make next buffer visible to everyone before the next iteration reads it
    asm volatile("s_wait_asynccnt 0x0" ::: "memory");
    __syncthreads();
  }

#pragma unroll
  for (int r = 0; r < 8; ++r) {
    const int row = p0 + hi * 8 + r;
    float v = acc[r] + bb;
    if (do_relu) v = v > 0.f ? v : 0.f;
    if (out_h) out_h[(size_t)row * N + n0 + ln] = (_Float16)v;
    else       out_f[(size_t)row * N + n0 + ln] = v;
  }
}

// ---------------------------------------------------------------------------
// Helpers: weight conversion/relayout, and pass-through tail outputs
// ---------------------------------------------------------------------------
__global__ void f32_to_f16_kernel(const float* __restrict__ src,
                                  _Float16* __restrict__ dst, int n)
{
  int i = blockIdx.x * blockDim.x + threadIdx.x;
  if (i < n) dst[i] = (_Float16)src[i];
}

// (E_out, E_in, KW) f32  ->  [tap][m][c] f16 (A-fragments contiguous in c)
__global__ void convw_to_f16_kernel(const float* __restrict__ w,
                                    _Float16* __restrict__ dst, int n)
{
  int i = blockIdx.x * blockDim.x + threadIdx.x;
  if (i < n) {
    const int tap = i % KW;
    const int c   = (i / KW) % E_;
    const int m   = i / (KW * E_);
    dst[((size_t)tap * E_ + m) * E_ + c] = (_Float16)w[i];
  }
}

__global__ void tail_copy_kernel(const float* __restrict__ trig,
                                 const int* __restrict__ labels,
                                 float* __restrict__ out)
{
  int i = blockIdx.x * blockDim.x + threadIdx.x;
  if (i < B_ * T_) out[(size_t)P_ * R_ + i] = trig[i];
  if (i < P_)      out[(size_t)P_ * R_ + B_ * T_ + i] = (float)labels[i];
}

// ---------------------------------------------------------------------------
extern "C" void kernel_launch(void* const* d_in, const int* in_sizes, int n_in,
                              void* d_out, int out_size, void* d_ws, size_t ws_size,
                              hipStream_t stream)
{
  (void)in_sizes; (void)n_in; (void)out_size; (void)ws_size;

  const float* x    = (const float*)d_in[0];
  const float* trig = (const float*)d_in[1];
  const int* bidx   = (const int*)d_in[2];
  const int* hidx   = (const int*)d_in[3];
  const int* hspan  = (const int*)d_in[4];
  const int* tidx   = (const int*)d_in[5];
  const int* tspan  = (const int*)d_in[6];
  const int* labels = (const int*)d_in[7];
  const float* c1w  = (const float*)d_in[8];
  const float* c1b  = (const float*)d_in[9];
  const float* c2w  = (const float*)d_in[10];
  const float* c2b  = (const float*)d_in[11];
  const float* w1   = (const float*)d_in[12];
  const float* b1   = (const float*)d_in[13];
  const float* w2   = (const float*)d_in[14];
  const float* b2   = (const float*)d_in[15];
  const float* wr   = (const float*)d_in[16];
  const float* br   = (const float*)d_in[17];
  float* out = (float*)d_out;

  // Workspace carve-up
  char* ws = (char*)d_ws;
  float*    x1    = (float*)ws;    ws += (size_t)B_ * E_ * T_ * 4;
  float*    x2    = (float*)ws;    ws += (size_t)B_ * E_ * T_ * 4;
  _Float16* maxed = (_Float16*)ws; ws += (size_t)P_ * E_ * 2;
  _Float16* h1    = (_Float16*)ws; ws += (size_t)P_ * H_ * 2;
  _Float16* h2    = (_Float16*)ws; ws += (size_t)P_ * H_ * 2;
  _Float16* w1h   = (_Float16*)ws; ws += (size_t)H_ * E_ * 2;
  _Float16* w2h   = (_Float16*)ws; ws += (size_t)H_ * H_ * 2;
  _Float16* wrh   = (_Float16*)ws; ws += (size_t)R_ * H_ * 2;
  _Float16* wc1h  = (_Float16*)ws; ws += (size_t)KW * E_ * E_ * 2;
  _Float16* wc2h  = (_Float16*)ws; ws += (size_t)KW * E_ * E_ * 2;

  // Weight precision/layout staging
  f32_to_f16_kernel<<<(H_ * E_ + 255) / 256, 256, 0, stream>>>(w1, w1h, H_ * E_);
  f32_to_f16_kernel<<<(H_ * H_ + 255) / 256, 256, 0, stream>>>(w2, w2h, H_ * H_);
  f32_to_f16_kernel<<<(R_ * H_ + 255) / 256, 256, 0, stream>>>(wr, wrh, R_ * H_);
  convw_to_f16_kernel<<<(KW * E_ * E_ + 255) / 256, 256, 0, stream>>>(
      c1w, wc1h, KW * E_ * E_);
  convw_to_f16_kernel<<<(KW * E_ * E_ + 255) / 256, 256, 0, stream>>>(
      c2w, wc2h, KW * E_ * E_);

  // Two residual convs (WMMA GEMMs)
  dim3 cgrid(T_ / 16, E_ / 128, B_);
  conv_res_kernel<<<cgrid, 256, 0, stream>>>(x,  wc1h, c1b, x1);
  conv_res_kernel<<<cgrid, 256, 0, stream>>>(x1, wc2h, c2b, x2);

  // Span-max gather -> f16 activations
  spanmax_kernel<<<P_, 256, 0, stream>>>(x2, bidx, hidx, hspan, tidx, tspan, maxed);

  // MLP: 512 -> 1024 -> 1024 -> 32 (last layer writes f32 logits to d_out)
  mlp_gemm_kernel<<<dim3(P_ / 16, H_ / 128), 256, 0, stream>>>(
      maxed, w1h, b1, h1, nullptr, E_, H_, 1);
  mlp_gemm_kernel<<<dim3(P_ / 16, H_ / 128), 256, 0, stream>>>(
      h1, w2h, b2, h2, nullptr, H_, H_, 1);
  mlp_gemm_kernel<<<dim3(P_ / 16, 1), 64, 0, stream>>>(
      h2, wrh, br, nullptr, out, H_, R_, 0);

  // Pass-through outputs
  tail_copy_kernel<<<32, 256, 0, stream>>>(trig, labels, out);
}
